// BiGCN_graphcl_78357383348239
// MI455X (gfx1250) — matrix operand order repaired and checked by reading, not compiled, not run on hardware
//
#include <hip/hip_runtime.h>
#include <hip/hip_bf16.h>

#define N_NODES   100000
#define E_EDGES   1600000
#define IN_FEATS  256
#define HIDDEN    128
#define NUM_GRAPHS 1024

typedef __attribute__((ext_vector_type(16))) __bf16 v16bf;
typedef __attribute__((ext_vector_type(8)))  float  v8f;

// ---------------------------------------------------------------------------
// Small utility kernels
// ---------------------------------------------------------------------------
__global__ void k_zero(float* __restrict__ p, int n) {
    int i = blockIdx.x * blockDim.x + threadIdx.x;
    if (i < n) p[i] = 0.0f;
}

__global__ void k_fill1_pair(float* __restrict__ a, float* __restrict__ b, int n) {
    int i = blockIdx.x * blockDim.x + threadIdx.x;
    if (i < n) { a[i] = 1.0f; b[i] = 1.0f; }   // self-loop contributes 1 to degree
}

// degree over dst (top-down branch) and over src (bottom-up branch)
__global__ void k_edge_deg(const int* __restrict__ src, const int* __restrict__ dst,
                           float* __restrict__ deg_td, float* __restrict__ deg_bu) {
    int i = blockIdx.x * blockDim.x + threadIdx.x;
    if (i < E_EDGES) {
        atomicAdd(&deg_td[dst[i]], 1.0f);
        atomicAdd(&deg_bu[src[i]], 1.0f);
    }
}

__global__ void k_rsqrt_pair(float* __restrict__ a, float* __restrict__ b, int n) {
    int i = blockIdx.x * blockDim.x + threadIdx.x;
    if (i < n) { a[i] = rsqrtf(a[i]); b[i] = rsqrtf(b[i]); }
}

// W [kin,kout] row-major f32  ->  Wt [kout,kin] row-major bf16 (B-fragment friendly)
__global__ void k_weight_t(const float* __restrict__ W, __bf16* __restrict__ Wt,
                           int kin, int kout) {
    int i = blockIdx.x * blockDim.x + threadIdx.x;
    if (i < kin * kout) {
        int r = i / kout, c = i % kout;
        Wt[c * kin + r] = (__bf16)W[i];
    }
}

// ---------------------------------------------------------------------------
// WMMA GEMM: C[N_NODES, HIDDEN] = A[N_NODES, KIN](f32) x Wt^T (Wt is [HIDDEN,KIN] bf16)
// 256 threads = 8 waves; each wave: one 16-row M-tile x 8 N-tiles of 16.
// Weights staged once per workgroup into LDS (padded rows -> conflict-free
// ds_load_b128: dword row stride = (KIN+8)/2 ≡ 4 (mod 64)).
// ---------------------------------------------------------------------------
template <int KIN>
__global__ __launch_bounds__(256) void k_gemm_wmma(const float* __restrict__ A,
                                                   const __bf16* __restrict__ Wt,
                                                   float* __restrict__ C) {
    constexpr int LSTRIDE = KIN + 8;                       // bf16 elems per padded row
    __shared__ __align__(16) __bf16 sW[HIDDEN * LSTRIDE];  // <= ~66 KB

    // ---- cooperative stage: Wt[HIDDEN][KIN] -> LDS, 16B chunks
    constexpr int CHUNKS = HIDDEN * KIN / 8;               // 8 bf16 per chunk
    for (int idx = threadIdx.x; idx < CHUNKS; idx += 256) {
        const int col = (idx * 8) / KIN;
        const int kof = (idx * 8) % KIN;
        *(uint4*)(&sW[col * LSTRIDE + kof]) = *(const uint4*)(Wt + idx * 8);
    }
    __syncthreads();                                       // before any wave can exit

    const int lane  = threadIdx.x & 31;
    const int wave  = threadIdx.x >> 5;
    const int mtile = blockIdx.x * 8 + wave;
    const int MT    = N_NODES / 16;                        // 6250, exact
    if (mtile >= MT) return;                               // wave-uniform guard

    const int m16 = lane & 15;
    const int hi  = lane >> 4;                             // 0: lanes 0-15, 1: 16-31
    const int row = mtile * 16 + m16;

    v8f acc[8];
#pragma unroll
    for (int t = 0; t < 8; ++t) acc[t] = (v8f)(0.0f);

    for (int k0 = 0; k0 < KIN; k0 += 32) {
        // ---- A fragment (16x32 bf16): lane holds K = kb..kb+15 contiguous,
        //      kb = k0 + 16*hi (matches B fragment order)
        const float* ap = A + (size_t)row * KIN + k0 + hi * 16;
        float4 a0 = *(const float4*)(ap);
        float4 a1 = *(const float4*)(ap + 4);
        float4 a2 = *(const float4*)(ap + 8);
        float4 a3 = *(const float4*)(ap + 12);
        v16bf afrag;
        afrag[0]  = (__bf16)a0.x; afrag[1]  = (__bf16)a0.y;
        afrag[2]  = (__bf16)a0.z; afrag[3]  = (__bf16)a0.w;
        afrag[4]  = (__bf16)a1.x; afrag[5]  = (__bf16)a1.y;
        afrag[6]  = (__bf16)a1.z; afrag[7]  = (__bf16)a1.w;
        afrag[8]  = (__bf16)a2.x; afrag[9]  = (__bf16)a2.y;
        afrag[10] = (__bf16)a2.z; afrag[11] = (__bf16)a2.w;
        afrag[12] = (__bf16)a3.x; afrag[13] = (__bf16)a3.y;
        afrag[14] = (__bf16)a3.z; afrag[15] = (__bf16)a3.w;

        // ---- B fragments (32x16 bf16) from LDS: lane = column,
        //      K = k0 + 16*hi .. +15 contiguous
        const int kb = k0 + hi * 16;
#pragma unroll
        for (int t = 0; t < 8; ++t) {
            const int col = t * 16 + m16;
            v16bf bfrag = *(const v16bf*)(&sW[col * LSTRIDE + kb]);
            acc[t] = __builtin_amdgcn_wmma_f32_16x16x32_bf16(
                false, afrag, false, bfrag, (short)0, acc[t], false, false);
        }
    }

    // ---- store D: VGPR r, lane l -> M = mtile*16 + r + 8*hi, N = t*16 + (l&15)
    const int rbase = mtile * 16 + hi * 8;
#pragma unroll
    for (int t = 0; t < 8; ++t) {
        const int col = t * 16 + m16;
#pragma unroll
        for (int r = 0; r < 8; ++r)
            C[(size_t)(rbase + r) * HIDDEN + col] = acc[t][r];
    }
}

// ---------------------------------------------------------------------------
// Edge propagation: acc[d] += h[s] * dinv[s]*dinv[d]   (one wave per edge;
// edge endpoints / norms forced scalar so they lower to SMEM loads; the only
// vector traffic is the 512B/wave feature gather + f32 atomics into L2)
// ---------------------------------------------------------------------------
__global__ __launch_bounds__(256) void k_edge_scatter(const float* __restrict__ h,
                                                      const int* __restrict__ s,
                                                      const int* __restrict__ d,
                                                      const float* __restrict__ dinv,
                                                      float* __restrict__ acc) {
    const int lane   = threadIdx.x & 31;
    const int wave   = (blockIdx.x * blockDim.x + threadIdx.x) >> 5;
    const int nwaves = (gridDim.x * blockDim.x) >> 5;
    for (int e = wave; e < E_EDGES; e += nwaves) {
        const int eu = __builtin_amdgcn_readfirstlane(e);   // wave-uniform edge id
        const int si = s[eu];
        const int di = d[eu];
        const float norm = dinv[si] * dinv[di];
        float4 v = ((const float4*)(h + (size_t)si * HIDDEN))[lane];
        float* o = acc + (size_t)di * HIDDEN + lane * 4;
        atomicAdd(o + 0, v.x * norm);
        atomicAdd(o + 1, v.y * norm);
        atomicAdd(o + 2, v.z * norm);
        atomicAdd(o + 3, v.w * norm);
    }
}

// self-loop + bias (+ReLU) fused, in place on acc
__global__ void k_post(float* __restrict__ acc, const float* __restrict__ h,
                       const float* __restrict__ dinv, const float* __restrict__ b,
                       int relu) {
    int i = blockIdx.x * blockDim.x + threadIdx.x;
    if (i >= N_NODES * HIDDEN) return;
    const int node = i >> 7, col = i & 127;
    const float dv = dinv[node];
    float v = acc[i] + h[i] * dv * dv + b[col];
    if (relu) v = fmaxf(v, 0.0f);
    acc[i] = v;
}

// global_add_pool into d_out[:, coloff:coloff+128]; node id is wave-uniform
// (128 consecutive threads share a node), so batch[] lowers to a scalar load.
__global__ void k_pool(const float* __restrict__ acc, const int* __restrict__ batch,
                       float* __restrict__ out, int coloff) {
    int i = blockIdx.x * blockDim.x + threadIdx.x;   // N*H = 12.8M, exact grid
    const int node = __builtin_amdgcn_readfirstlane(i >> 7);
    const int col  = i & 127;
    atomicAdd(&out[(size_t)batch[node] * (2 * HIDDEN) + coloff + col],
              acc[(size_t)node * HIDDEN + col]);
}

// ---------------------------------------------------------------------------
extern "C" void kernel_launch(void* const* d_in, const int* in_sizes, int n_in,
                              void* d_out, int out_size, void* d_ws, size_t ws_size,
                              hipStream_t stream) {
    const float* x     = (const float*)d_in[0];
    const int*   ei    = (const int*)d_in[1];
    const int*   src   = ei;
    const int*   dst   = ei + E_EDGES;
    const int*   batch = (const int*)d_in[2];
    const float* W_td1 = (const float*)d_in[3];
    const float* b_td1 = (const float*)d_in[4];
    const float* W_td2 = (const float*)d_in[5];
    const float* b_td2 = (const float*)d_in[6];
    const float* W_bu1 = (const float*)d_in[7];
    const float* b_bu1 = (const float*)d_in[8];
    const float* W_bu2 = (const float*)d_in[9];
    const float* b_bu2 = (const float*)d_in[10];

    // ---- workspace carve-out
    char* ws = (char*)d_ws;
    float* h   = (float*)ws;  ws += (size_t)N_NODES * HIDDEN * sizeof(float);   // 51.2 MB
    float* acc = (float*)ws;  ws += (size_t)N_NODES * HIDDEN * sizeof(float);   // 51.2 MB
    float* dinv_td = (float*)ws; ws += (size_t)N_NODES * sizeof(float);
    float* dinv_bu = (float*)ws; ws += (size_t)N_NODES * sizeof(float);
    __bf16* Wt1_td = (__bf16*)ws; ws += (size_t)IN_FEATS * HIDDEN * sizeof(__bf16);
    __bf16* Wt2_td = (__bf16*)ws; ws += (size_t)HIDDEN   * HIDDEN * sizeof(__bf16);
    __bf16* Wt1_bu = (__bf16*)ws; ws += (size_t)IN_FEATS * HIDDEN * sizeof(__bf16);
    __bf16* Wt2_bu = (__bf16*)ws; ws += (size_t)HIDDEN   * HIDDEN * sizeof(__bf16);

    const int TB = 256;
    const int NH = N_NODES * HIDDEN;
    const int nh_blocks   = (NH + TB - 1) / TB;          // 50000 exact
    const int gemm_blocks = (N_NODES / 16 + 7) / 8;      // 782

    // ---- prep: output zero, degrees, dinv, transposed bf16 weights
    k_zero<<<(NUM_GRAPHS * 2 * HIDDEN + TB - 1) / TB, TB, 0, stream>>>((float*)d_out,
                                                                        NUM_GRAPHS * 2 * HIDDEN);
    k_fill1_pair<<<(N_NODES + TB - 1) / TB, TB, 0, stream>>>(dinv_td, dinv_bu, N_NODES);
    k_edge_deg<<<(E_EDGES + TB - 1) / TB, TB, 0, stream>>>(src, dst, dinv_td, dinv_bu);
    k_rsqrt_pair<<<(N_NODES + TB - 1) / TB, TB, 0, stream>>>(dinv_td, dinv_bu, N_NODES);
    k_weight_t<<<(IN_FEATS * HIDDEN + TB - 1) / TB, TB, 0, stream>>>(W_td1, Wt1_td, IN_FEATS, HIDDEN);
    k_weight_t<<<(HIDDEN   * HIDDEN + TB - 1) / TB, TB, 0, stream>>>(W_td2, Wt2_td, HIDDEN, HIDDEN);
    k_weight_t<<<(IN_FEATS * HIDDEN + TB - 1) / TB, TB, 0, stream>>>(W_bu1, Wt1_bu, IN_FEATS, HIDDEN);
    k_weight_t<<<(HIDDEN   * HIDDEN + TB - 1) / TB, TB, 0, stream>>>(W_bu2, Wt2_bu, HIDDEN, HIDDEN);

    // ---- one GCN branch (stream-ordered; h/acc ping-pong)
    auto branch = [&](const int* s, const int* d, const float* dinv,
                      const __bf16* Wt1, const float* b1,
                      const __bf16* Wt2, const float* b2, int coloff) {
        // layer 1
        k_gemm_wmma<IN_FEATS><<<gemm_blocks, TB, 0, stream>>>(x, Wt1, h);
        k_zero<<<nh_blocks, TB, 0, stream>>>(acc, NH);
        k_edge_scatter<<<2048, TB, 0, stream>>>(h, s, d, dinv, acc);
        k_post<<<nh_blocks, TB, 0, stream>>>(acc, h, dinv, b1, /*relu=*/1);
        // layer 2 (reads acc, writes h; acc reused as accumulator afterwards)
        k_gemm_wmma<HIDDEN><<<gemm_blocks, TB, 0, stream>>>(acc, Wt2, h);
        k_zero<<<nh_blocks, TB, 0, stream>>>(acc, NH);
        k_edge_scatter<<<2048, TB, 0, stream>>>(h, s, d, dinv, acc);
        k_post<<<nh_blocks, TB, 0, stream>>>(acc, h, dinv, b2, /*relu=*/0);
        k_pool<<<nh_blocks, TB, 0, stream>>>(acc, batch, (float*)d_out, coloff);
    };

    branch(src, dst, dinv_td, Wt1_td, b_td1, Wt2_td, b_td2, 0);       // top-down
    branch(dst, src, dinv_bu, Wt1_bu, b_bu1, Wt2_bu, b_bu2, HIDDEN);  // bottom-up
}
